// PeepholeLSTMCell_13649406066962
// MI455X (gfx1250) — matrix operand order repaired and checked by reading, not compiled, not run on hardware
//
#include <hip/hip_runtime.h>
#include <cstdint>

typedef _Float16 v16h __attribute__((ext_vector_type(16)));
typedef _Float16 v8h  __attribute__((ext_vector_type(8)));
typedef _Float16 v4h  __attribute__((ext_vector_type(4)));
typedef float    v8f  __attribute__((ext_vector_type(8)));

#define B_ROWS   16384
#define IN_DIM   512
#define H_DIM    512
#define NG       2048        // 4*H
#define M_BLK    64          // rows per block
#define H_BLK    64          // h columns per block (per gate)
#define NCOLS    256         // 4 gates * H_BLK
#define THREADS  512         // 16 wave32 waves
#define KSTEP    32
#define NKK      32          // (IN + H) / KSTEP

// LDS layout (bytes):
//  [0,      65536)  xs : 64x512 f16  (reused as gates 64x256 f32 after GEMM)
//  [65536, 131072)  hs : 64x512 f16
//  [131072,147456)  wbuf0 : 256 cols x 32 k f16 (16 KB)
//  [147456,163840)  wbuf1
#define LDS_BYTES   (160 * 1024)
#define WBUF0_OFF   131072
#define WBUF1_OFF   147456

__device__ __forceinline__ float sigmoid_fast(float x) {
  return __frcp_rn(1.0f + __expf(-x));
}
__device__ __forceinline__ float tanh_fast(float x) {
  float e = __expf(-2.0f * x);
  return (1.0f - e) * __frcp_rn(1.0f + e);
}

// ---- async weight staging: f16 weights (ws) -> LDS, bypassing VGPRs ----
// tile = 256 block-cols x 32 k halves = 16 KB; 512 threads x 32 B each.
__device__ __forceinline__ void stage_async(const _Float16* wsrc, int klocal,
                                            unsigned ldsByte, int tid, int byId) {
  const int c   = tid >> 1;        // block column 0..255
  const int seg = tid & 1;         // which 32B half of this column's 64B chunk
  const int g   = c >> 6;
  const int hl  = c & 63;
  const int nrow = g * H_DIM + byId * H_BLK + hl;     // row in W [2048 x 512]
  const unsigned voff = (unsigned)(nrow * (IN_DIM * 2) + klocal * 2 + seg * 32);
  const unsigned lds  = ldsByte + (unsigned)(c * 64 + seg * 32);
  asm volatile(
      "global_load_async_to_lds_b128 %0, %1, %2 offset:0\n\t"
      "global_load_async_to_lds_b128 %0, %1, %2 offset:16"
      :: "v"(lds), "v"(voff), "s"((unsigned long long)(uintptr_t)wsrc)
      : "memory");
}

// ---- fallback staging: fp32 weights -> convert in VGPRs -> ds_store ----
__device__ __forceinline__ void stage_vgpr(const float* wsrc, int klocal,
                                           _Float16* ldsDst, int tid, int byId) {
  const int c   = tid >> 1;
  const int seg = tid & 1;
  const int g   = c >> 6;
  const int hl  = c & 63;
  const int nrow = g * H_DIM + byId * H_BLK + hl;
  const float4* s4 = (const float4*)(wsrc + (size_t)nrow * IN_DIM + klocal + seg * 16);
  const float4 f0 = s4[0], f1 = s4[1], f2 = s4[2], f3 = s4[3];
  v8h lo, hi;
  lo[0] = (_Float16)f0.x; lo[1] = (_Float16)f0.y; lo[2] = (_Float16)f0.z; lo[3] = (_Float16)f0.w;
  lo[4] = (_Float16)f1.x; lo[5] = (_Float16)f1.y; lo[6] = (_Float16)f1.z; lo[7] = (_Float16)f1.w;
  hi[0] = (_Float16)f2.x; hi[1] = (_Float16)f2.y; hi[2] = (_Float16)f2.z; hi[3] = (_Float16)f2.w;
  hi[4] = (_Float16)f3.x; hi[5] = (_Float16)f3.y; hi[6] = (_Float16)f3.z; hi[7] = (_Float16)f3.w;
  _Float16* d = ldsDst + c * KSTEP + seg * 16;
  *(v8h*)(d + 0) = lo;
  *(v8h*)(d + 8) = hi;
}

// ---- one K=32 step: load A + all 4 B fragments, then 4 back-to-back WMMAs ----
__device__ __forceinline__ void compute_step(v8f acc[4], const _Float16* aTile, int kb,
                                             const _Float16* wtile, int mg, int g, int lane) {
  const int row16 = lane & 15;
  const int hi    = (lane >> 4) & 1;
  const int arow  = mg * 16 + row16;

  // A fragment (16x32 f16): lanes 0-15 K=0-7/16-23, lanes 16-31 K=8-15/24-31
  const v8h alo = *(const v8h*)(aTile + arow * IN_DIM + kb + hi * 8);
  const v8h ahi = *(const v8h*)(aTile + arow * IN_DIM + kb + 16 + hi * 8);
  const v16h a = __builtin_shufflevector(alo, ahi,
      0, 1, 2, 3, 4, 5, 6, 7, 8, 9, 10, 11, 12, 13, 14, 15);

  v16h b[4];
#pragma unroll
  for (int t = 0; t < 4; ++t) {
    const int c = g * H_BLK + t * 16 + row16;      // B fragment: lane = column
    const v8h blo = *(const v8h*)(wtile + c * KSTEP + hi * 8);
    const v8h bhi = *(const v8h*)(wtile + c * KSTEP + 16 + hi * 8);
    b[t] = __builtin_shufflevector(blo, bhi,
        0, 1, 2, 3, 4, 5, 6, 7, 8, 9, 10, 11, 12, 13, 14, 15);
  }
#pragma unroll
  for (int t = 0; t < 4; ++t) {
    acc[t] = __builtin_amdgcn_wmma_f32_16x16x32_f16(
        false, a, false, b[t], (short)0, acc[t], false, false);
  }
}

template <bool WF16>
__global__ void __launch_bounds__(THREADS)
lstm_peephole_fused(const float* __restrict__ x,  const float* __restrict__ hx,
                    const float* __restrict__ cx, const void* __restrict__ WxP,
                    const float* __restrict__ bx, const void* __restrict__ WhP,
                    const float* __restrict__ bh, const float* __restrict__ w_ci,
                    const float* __restrict__ w_cf, const float* __restrict__ w_co,
                    float* __restrict__ out) {
  __shared__ __align__(16) unsigned char smem[LDS_BYTES];
  _Float16* xs = (_Float16*)smem;                   // [64][512] f16
  _Float16* hs = xs + M_BLK * IN_DIM;               // [64][512] f16
  _Float16* wb0 = (_Float16*)(smem + WBUF0_OFF);
  _Float16* wb1 = (_Float16*)(smem + WBUF1_OFF);
  float* gbuf = (float*)smem;                       // [64][256] f32 (reuses xs)

  const int tid  = threadIdx.x;
  const int byId = blockIdx.y;                      // h-column block (0..7)
  const int rowBase = blockIdx.x * M_BLK;

  // ---- kick off weight chunk 0 staging into wb0 ----
  if (WF16) {
    stage_async((const _Float16*)WxP, 0, WBUF0_OFF, tid, byId);
  } else {
    stage_vgpr((const float*)WxP, 0, wb0, tid, byId);
  }

  // ---- stage x, hx tiles -> f16 LDS ----
  {
    const float4* x4 = (const float4*)(x + (size_t)rowBase * IN_DIM);
    const float4* h4 = (const float4*)(hx + (size_t)rowBase * IN_DIM);
    for (int i = tid; i < (M_BLK * IN_DIM) / 4; i += THREADS) {
      const float4 vx = x4[i];
      const float4 vh = h4[i];
      v4h cvx = { (_Float16)vx.x, (_Float16)vx.y, (_Float16)vx.z, (_Float16)vx.w };
      v4h cvh = { (_Float16)vh.x, (_Float16)vh.y, (_Float16)vh.z, (_Float16)vh.w };
      *(v4h*)(xs + i * 4) = cvx;
      *(v4h*)(hs + i * 4) = cvh;
    }
  }

  const int wave = tid >> 5;
  const int lane = tid & 31;
  const int mg   = wave & 3;   // m-group: rows [mg*16, +16)
  const int g    = wave >> 2;  // gate index 0..3

  v8f acc[4] = {};

  // ---- double-buffered GEMM loop, manually unrolled by 2 so each half has a
  //      fixed weight buffer (no runtime select -> no accumulator copies) ----
  for (int kk = 0; kk < NKK; kk += 2) {
    // -- half A: consume wb0, stage kk+1 into wb1 --
    {
      const int j = kk + 1;  // always < NKK
      const int klocal = (j & 15) * KSTEP;
      if (WF16) {
        const _Float16* wsrc = (j < 16) ? (const _Float16*)WxP : (const _Float16*)WhP;
        stage_async(wsrc, klocal, WBUF1_OFF, tid, byId);
        asm volatile("s_wait_asynccnt 0x2" ::: "memory");
      } else {
        const float* wsrc = (j < 16) ? (const float*)WxP : (const float*)WhP;
        stage_vgpr(wsrc, klocal, wb1, tid, byId);
      }
    }
    __syncthreads();
    {
      const _Float16* aT = (kk < 16) ? xs : hs;
      compute_step(acc, aT, (kk & 15) * KSTEP, wb0, mg, g, lane);
    }
    __syncthreads();

    // -- half B: consume wb1, stage kk+2 into wb0 --
    if (kk + 2 < NKK) {
      const int j = kk + 2;
      const int klocal = (j & 15) * KSTEP;
      if (WF16) {
        const _Float16* wsrc = (j < 16) ? (const _Float16*)WxP : (const _Float16*)WhP;
        stage_async(wsrc, klocal, WBUF0_OFF, tid, byId);
        asm volatile("s_wait_asynccnt 0x2" ::: "memory");
      } else {
        const float* wsrc = (j < 16) ? (const float*)WxP : (const float*)WhP;
        stage_vgpr(wsrc, klocal, wb0, tid, byId);
      }
    } else {
      if (WF16) asm volatile("s_wait_asynccnt 0x0" ::: "memory");
    }
    __syncthreads();
    {
      const int j = kk + 1;
      const _Float16* aT = (j < 16) ? xs : hs;
      compute_step(acc, aT, (j & 15) * KSTEP, wb1, mg, g, lane);
    }
    __syncthreads();
  }

  // ---- exchange gate pre-activations through LDS (reuse xs region) ----
  {
    const int row16 = lane & 15;
    const int mBase = ((lane >> 4) & 1) * 8;
#pragma unroll
    for (int t = 0; t < 4; ++t) {
      const int c = g * H_BLK + t * 16 + row16;
#pragma unroll
      for (int r = 0; r < 8; ++r)
        gbuf[(mg * 16 + mBase + r) * NCOLS + c] = acc[t][r];
    }
  }
  __syncthreads();

  // ---- fused bias + peephole + activations + store ----
  float* out_h = out;
  float* out_c = out + (size_t)B_ROWS * H_DIM;
  for (int i = tid; i < M_BLK * H_BLK; i += THREADS) {
    const int row = i >> 6;
    const int hl  = i & (H_BLK - 1);
    const int hgl = byId * H_BLK + hl;
    const int b   = rowBase + row;

    const float ip = gbuf[row * NCOLS + 0 * H_BLK + hl] + bx[0 * H_DIM + hgl] + bh[0 * H_DIM + hgl];
    const float fp = gbuf[row * NCOLS + 1 * H_BLK + hl] + bx[1 * H_DIM + hgl] + bh[1 * H_DIM + hgl];
    const float gp = gbuf[row * NCOLS + 2 * H_BLK + hl] + bx[2 * H_DIM + hgl] + bh[2 * H_DIM + hgl];
    const float op = gbuf[row * NCOLS + 3 * H_BLK + hl] + bx[3 * H_DIM + hgl] + bh[3 * H_DIM + hgl];

    const float c  = cx[(size_t)b * H_DIM + hgl];
    const float ig = sigmoid_fast(ip + c * w_ci[hgl]);
    const float fg = sigmoid_fast(fp + c * w_cf[hgl]);
    const float gg = tanh_fast(gp);
    const float cn = fg * c + ig * gg;
    const float og = sigmoid_fast(op + cn * w_co[hgl]);

    out_h[(size_t)b * H_DIM + hgl] = og * tanh_fast(cn);
    out_c[(size_t)b * H_DIM + hgl] = cn;
  }
}

// ---- one-shot fp32 -> f16 weight conversion into workspace ----
__global__ void __launch_bounds__(THREADS)
convert_weights(const float* __restrict__ Wx, const float* __restrict__ Wh,
                _Float16* __restrict__ dst) {
  const int i4 = blockIdx.x * THREADS + threadIdx.x;   // float4 index
  const int NW4 = (NG * IN_DIM) / 4;                   // 262144 per matrix
  if (i4 >= 2 * NW4) return;
  const float4 v = (i4 < NW4) ? ((const float4*)Wx)[i4]
                              : ((const float4*)Wh)[i4 - NW4];
  v4h h = { (_Float16)v.x, (_Float16)v.y, (_Float16)v.z, (_Float16)v.w };
  ((v4h*)dst)[i4] = h;
}

extern "C" void kernel_launch(void* const* d_in, const int* in_sizes, int n_in,
                              void* d_out, int out_size, void* d_ws, size_t ws_size,
                              hipStream_t stream) {
  (void)in_sizes; (void)n_in; (void)out_size;
  const float* x    = (const float*)d_in[0];
  const float* hx   = (const float*)d_in[1];
  const float* cx   = (const float*)d_in[2];
  const float* Wx   = (const float*)d_in[3];
  const float* bx   = (const float*)d_in[4];
  const float* Wh   = (const float*)d_in[5];
  const float* bh   = (const float*)d_in[6];
  const float* w_ci = (const float*)d_in[7];
  const float* w_cf = (const float*)d_in[8];
  const float* w_co = (const float*)d_in[9];
  float* out = (float*)d_out;

  dim3 grid(B_ROWS / M_BLK, H_DIM / H_BLK);  // 256 x 8
  dim3 block(THREADS);

  const size_t needW = (size_t)2 * NG * IN_DIM * sizeof(_Float16);  // 4 MB
  if (ws_size >= needW) {
    _Float16* Wx16 = (_Float16*)d_ws;
    _Float16* Wh16 = Wx16 + (size_t)NG * IN_DIM;
    convert_weights<<<(2 * (NG * IN_DIM / 4) + THREADS - 1) / THREADS, THREADS, 0, stream>>>(
        Wx, Wh, Wx16);
    lstm_peephole_fused<true><<<grid, block, 0, stream>>>(
        x, hx, cx, (const void*)Wx16, bx, (const void*)Wh16, bh, w_ci, w_cf, w_co, out);
  } else {
    lstm_peephole_fused<false><<<grid, block, 0, stream>>>(
        x, hx, cx, (const void*)Wx, bx, (const void*)Wh, bh, w_ci, w_cf, w_co, out);
  }
}
